// TritonFusedRCMAttention_89309549953027
// MI455X (gfx1250) — matrix operand order repaired and checked by reading, not compiled, not run on hardware
//
#include <hip/hip_runtime.h>
#include <hip/hip_bf16.h>

typedef _Float16 half_t;
typedef __attribute__((ext_vector_type(16))) _Float16 v16h;
typedef __attribute__((ext_vector_type(8)))  float    v8f;
typedef __attribute__((ext_vector_type(4)))  float    v4f;

__device__ __forceinline__ v8f wmma_f16(v16h a, v16h b, v8f c) {
  return __builtin_amdgcn_wmma_f32_16x16x32_f16(false, a, false, b, (short)0, c, false, false);
}

// 16B global -> LDS async copy (ASYNCcnt-tracked, bypasses VGPRs).
// ldsOff = low 32 bits of the generic pointer to __shared__ (flat->LDS trunc).
__device__ __forceinline__ void async_copy_b128(void* lds, const void* g) {
  unsigned ldsOff = (unsigned)(size_t)lds;
  unsigned long long ga = (unsigned long long)(size_t)g;
  asm volatile("global_load_async_to_lds_b128 %0, %1, off"
               :: "v"(ldsOff), "v"(ga) : "memory");
}
__device__ __forceinline__ void async_wait0() {
  asm volatile("s_wait_asynccnt 0x0" ::: "memory");
}

// ---------------------------------------------------------------------------
// K1: fp32 -> f16 conversion (vectorized 4/thread, NT reads)
// ---------------------------------------------------------------------------
__global__ __launch_bounds__(256) void cvt_f32_f16(const float* __restrict__ src,
                                                   half_t* __restrict__ dst, int n4) {
  int i = blockIdx.x * blockDim.x + threadIdx.x;
  if (i < n4) {
    v4f v = __builtin_nontemporal_load((const v4f*)src + i);
    half_t h[4] = {(half_t)v[0], (half_t)v[1], (half_t)v[2], (half_t)v[3]};
    ((uint2*)dst)[i] = *(const uint2*)h;
  }
}

// ---------------------------------------------------------------------------
// K2 / K5: tiled WMMA GEMM.  C[M x Nout] = A[M x K] * W[K x Nout] + bias
// Block tile 64x64, 128 threads (4 waves), K-step 32.
// A tile [64 x 32] async-staged (contiguous A-frag reads).
// W tile [32 x 64] staged transposed via VGPRs (contiguous B-frag reads).
// mode 0: write fp32 (NT) to outF (proj GEMM)
// mode 1: write f16 split into qkv layout [3][B][H][Nseq][64]
// ---------------------------------------------------------------------------
#define A_STRIDE  48   // halves per A row (96B, 16B-multiple for async chunks)
#define WT_STRIDE 34   // halves per transposed-W row

__global__ __launch_bounds__(128) void gemm_wmma(const half_t* __restrict__ A,
                                                 const half_t* __restrict__ W,
                                                 const float* __restrict__ bias,
                                                 float* __restrict__ outF,
                                                 half_t* __restrict__ qkv,
                                                 int M, int K, int Nout,
                                                 int mode, int B, int Nseq) {
  __shared__ half_t aLds[64 * A_STRIDE];    // [row][k]
  __shared__ half_t wLds[64 * WT_STRIDE];   // [n][k] (transposed)

  const int tid  = threadIdx.x;
  const int w    = tid >> 5;
  const int lane = tid & 31;
  const int m    = lane & 15;
  const int hi   = lane >> 4;
  const int rowBase = blockIdx.y * 64;
  const int colBase = blockIdx.x * 64;

  v8f acc[4];
#pragma unroll
  for (int j = 0; j < 4; j++)
#pragma unroll
    for (int r = 0; r < 8; r++) acc[j][r] = 0.0f;

  for (int k0 = 0; k0 < K; k0 += 32) {
    __syncthreads();   // previous iteration's fragment reads complete
    // A tile: 64 rows x 32 halves = 256 x 16B chunks, 2 per thread (async)
#pragma unroll
    for (int i = 0; i < 2; i++) {
      int c = tid * 2 + i;
      int r = c >> 2, ch = (c & 3) * 8;            // 8 halves per chunk
      async_copy_b128(&aLds[r * A_STRIDE + ch],
                      A + (size_t)(rowBase + r) * K + k0 + ch);
    }
    // W tile: read [k][n] coalesced, write transposed [n][k]
    {
      int kk = tid >> 2;
      int n0 = (tid & 3) * 16;
      const unsigned int* g = (const unsigned int*)(W + (size_t)(k0 + kk) * Nout + colBase + n0);
#pragma unroll
      for (int i = 0; i < 8; i++) {
        unsigned int d = g[i];
        int n = n0 + 2 * i;
        wLds[(n + 0) * WT_STRIDE + kk] = ((const half_t*)&d)[0];
        wLds[(n + 1) * WT_STRIDE + kk] = ((const half_t*)&d)[1];
      }
    }
    async_wait0();
    __syncthreads();

    v16h aF;
    {   // A fragment: lane m = row, halves K = 16*hi + e (contiguous)
      const half_t* base = &aLds[(w * 16 + m) * A_STRIDE];
#pragma unroll
      for (int e = 0; e < 16; e++) {
        int k = 16 * hi + e;
        aF[e] = base[k];
      }
    }
#pragma unroll
    for (int j = 0; j < 4; j++) {
      v16h bF;   // B fragment: lane m = col n, halves K = 16*hi + e (contiguous)
      const half_t* base = &wLds[(16 * j + m) * WT_STRIDE + 16 * hi];
#pragma unroll
      for (int e = 0; e < 16; e++) bF[e] = base[e];
      acc[j] = wmma_f16(aF, bF, acc[j]);
    }
  }

  // epilogue: C element r -> row r + 8*hi, col = lane&15
#pragma unroll
  for (int j = 0; j < 4; j++) {
#pragma unroll
    for (int r = 0; r < 8; r++) {
      int row = rowBase + w * 16 + r + 8 * hi;
      int col = colBase + 16 * j + m;
      float v = acc[j][r] + bias[col];
      if (mode == 0) {
        __builtin_nontemporal_store(v, &outF[(size_t)row * Nout + col]);
      } else {
        int s = col >> 10, rem = col & 1023;
        int h = rem >> 6,  d   = rem & 63;
        int b = row / Nseq, n  = row % Nseq;
        size_t idx = ((((size_t)s * B + b) * 16 + h) * (size_t)Nseq + n) * 64 + d;
        qkv[idx] = (half_t)v;
      }
    }
  }
}

// ---------------------------------------------------------------------------
// K3: flash attention per (b,h).  Block = 128 threads (4 waves), each wave
// owns 16 query rows; key tiles of 32; online softmax in fp32.
// K tile async-staged in memory order (contiguous kF reads);
// V tile staged transposed via VGPRs (contiguous vF reads).
// ---------------------------------------------------------------------------
#define K_STRIDE  80   // halves per key row (160B, 16B-multiple)
#define VT_STRIDE 34   // halves per transposed-V row

__global__ __launch_bounds__(128) void flash_attn(const half_t* __restrict__ qkv,
                                                  half_t* __restrict__ oh,
                                                  int B, int Nseq) {
  __shared__ half_t kB[32 * K_STRIDE];    // [key][dim]
  __shared__ half_t vT[64 * VT_STRIDE];   // [dim][key] (transposed)
  __shared__ half_t pB[4][16 * 34];       // per-wave P staging [row][key]

  const int tid  = threadIdx.x;
  const int w    = tid >> 5;
  const int lane = tid & 31;
  const int m    = lane & 15;
  const int hi   = lane >> 4;
  const int h    = blockIdx.y;
  const int b    = blockIdx.z;
  const int qbase = blockIdx.x * 64 + w * 16;

  const size_t S = (size_t)B * 16 * Nseq * 64;
  const size_t headBase = (((size_t)b * 16 + h) * (size_t)Nseq) * 64;
  const half_t* qg = qkv + 0 * S + headBase;
  const half_t* kg = qkv + 1 * S + headBase;
  const half_t* vg = qkv + 2 * S + headBase;

  // q fragments: two 16x32 A-frags covering Dh=64
  v16h qF[2];
#pragma unroll
  for (int f = 0; f < 2; f++) {
    const half_t* base = qg + (size_t)(qbase + m) * 64 + 32 * f;
#pragma unroll
    for (int e = 0; e < 16; e++) {
      int k = 16 * hi + e;
      qF[f][e] = base[k];
    }
  }

  float mrow[8], lrow[8];
  v8f O[4];
#pragma unroll
  for (int r = 0; r < 8; r++) { mrow[r] = -1e30f; lrow[r] = 0.0f; }
#pragma unroll
  for (int j = 0; j < 4; j++)
#pragma unroll
    for (int r = 0; r < 8; r++) O[j][r] = 0.0f;

  const float scale = 0.125f;   // 64^-0.5

  for (int n0 = 0; n0 < Nseq; n0 += 32) {
    if (n0 + 32 < Nseq) {       // gfx1250 global_prefetch_b8 on next tiles
      __builtin_prefetch(kg + (size_t)(n0 + 32) * 64, 0, 1);
      __builtin_prefetch(vg + (size_t)(n0 + 32) * 64, 0, 1);
    }
    __syncthreads();
    // K tile: 32 keys x 64 halves = 256 x 16B chunks, 2 per thread (async)
#pragma unroll
    for (int i = 0; i < 2; i++) {
      int c = tid * 2 + i;
      int key = c >> 3, ch = (c & 7) * 8;
      async_copy_b128(&kB[key * K_STRIDE + ch],
                      kg + (size_t)(n0 + key) * 64 + ch);
    }
    // V tile: read [key][dim] coalesced, write transposed [dim][key]
    {
      int key = tid >> 2, d0 = (tid & 3) * 16;
      const unsigned int* g = (const unsigned int*)(vg + (size_t)(n0 + key) * 64 + d0);
#pragma unroll
      for (int i = 0; i < 8; i++) {
        unsigned int dte = g[i];
        int d = d0 + 2 * i;
        vT[(d + 0) * VT_STRIDE + key] = ((const half_t*)&dte)[0];
        vT[(d + 1) * VT_STRIDE + key] = ((const half_t*)&dte)[1];
      }
    }
    async_wait0();
    __syncthreads();

    // S = q @ k^T  (two 16x16 C tiles covering 32 keys)
    v8f Sg[2];
#pragma unroll
    for (int g2 = 0; g2 < 2; g2++) {
#pragma unroll
      for (int r = 0; r < 8; r++) Sg[g2][r] = 0.0f;
      const half_t* kb = &kB[(g2 * 16 + m) * K_STRIDE];
#pragma unroll
      for (int f = 0; f < 2; f++) {
        v16h kF;   // element (kk=dim, n=key): contiguous dims per lane
#pragma unroll
        for (int e = 0; e < 16; e++) kF[e] = kb[32 * f + 16 * hi + e];
        Sg[g2] = wmma_f16(qF[f], kF, Sg[g2]);
      }
#pragma unroll
      for (int r = 0; r < 8; r++) Sg[g2][r] *= scale;
    }

    // online softmax: row stats across the 16-lane column groups
    float tmax[8];
#pragma unroll
    for (int r = 0; r < 8; r++) tmax[r] = fmaxf(Sg[0][r], Sg[1][r]);
#pragma unroll
    for (int mask = 1; mask <= 8; mask <<= 1)
#pragma unroll
      for (int r = 0; r < 8; r++) tmax[r] = fmaxf(tmax[r], __shfl_xor(tmax[r], mask, 32));

    float corr[8];
#pragma unroll
    for (int r = 0; r < 8; r++) {
      float nm = fmaxf(mrow[r], tmax[r]);
      corr[r] = __expf(mrow[r] - nm);
      mrow[r] = nm;
    }
#pragma unroll
    for (int r = 0; r < 8; r++) {
      Sg[0][r] = __expf(Sg[0][r] - mrow[r]);
      Sg[1][r] = __expf(Sg[1][r] - mrow[r]);
    }
    float rsum[8];
#pragma unroll
    for (int r = 0; r < 8; r++) rsum[r] = Sg[0][r] + Sg[1][r];
#pragma unroll
    for (int mask = 1; mask <= 8; mask <<= 1)
#pragma unroll
      for (int r = 0; r < 8; r++) rsum[r] += __shfl_xor(rsum[r], mask, 32);
#pragma unroll
    for (int r = 0; r < 8; r++) lrow[r] = lrow[r] * corr[r] + rsum[r];
#pragma unroll
    for (int j = 0; j < 4; j++)
#pragma unroll
      for (int r = 0; r < 8; r++) O[j][r] *= corr[r];

    // P (C-layout) -> per-wave LDS -> A-layout fragment
    half_t* P = &pB[w][0];
#pragma unroll
    for (int g2 = 0; g2 < 2; g2++)
#pragma unroll
      for (int r = 0; r < 8; r++)
        P[(r + 8 * hi) * 34 + g2 * 16 + m] = (half_t)Sg[g2][r];
    __builtin_amdgcn_wave_barrier();   // same-wave LDS ops are in-order

    v16h pF;
    {
      const half_t* pb = &P[m * 34];
#pragma unroll
      for (int e = 0; e < 16; e++) {
        int k = 16 * hi + e;
        pF[e] = pb[k];
      }
    }
#pragma unroll
    for (int j = 0; j < 4; j++) {
      v16h vF;   // B-frag of V: lane m = dim col, halves = key K = 16*hi+e (contiguous)
      const half_t* vb = &vT[(16 * j + m) * VT_STRIDE + 16 * hi];
#pragma unroll
      for (int e = 0; e < 16; e++) vF[e] = vb[e];
      O[j] = wmma_f16(pF, vF, O[j]);
    }
  }

  // normalize and write o as f16 in [B, N, C] layout (col = h*64 + dim)
#pragma unroll
  for (int j = 0; j < 4; j++) {
#pragma unroll
    for (int r = 0; r < 8; r++) {
      int row = qbase + r + 8 * hi;
      int col = h * 64 + 16 * j + m;
      float v = O[j][r] / lrow[r];
      oh[((size_t)b * Nseq + row) * 1024 + col] = (half_t)v;
    }
  }
}

// ---------------------------------------------------------------------------
// K4: u = c_skip * x + c_out * o   (SIGMA_DATA = 1), output f16
// ---------------------------------------------------------------------------
__global__ __launch_bounds__(256) void epilogue_u(const float* __restrict__ x,
                                                  const half_t* __restrict__ oh,
                                                  const float* __restrict__ t,
                                                  half_t* __restrict__ u,
                                                  int B, int n) {
  int i = blockIdx.x * blockDim.x + threadIdx.x;
  if (i >= n) return;
  float sk = 0.0f, so = 0.0f;
  for (int b = 0; b < B; b++) {
    float tv = t[b];
    float d = 1.0f + tv * tv;
    sk += 1.0f / d;
    so += tv * rsqrtf(d);
  }
  sk /= (float)B;
  so /= (float)B;
  u[i] = (half_t)(sk * x[i] + so * (float)oh[i]);
}

// ---------------------------------------------------------------------------
// launch
// ---------------------------------------------------------------------------
extern "C" void kernel_launch(void* const* d_in, const int* in_sizes, int n_in,
                              void* d_out, int out_size, void* d_ws, size_t ws_size,
                              hipStream_t stream) {
  const float* x     = (const float*)d_in[0];
  const float* t     = (const float*)d_in[1];
  const float* Wqkv  = (const float*)d_in[2];
  const float* bqkv  = (const float*)d_in[3];
  const float* Wproj = (const float*)d_in[4];
  const float* bproj = (const float*)d_in[5];
  float* out = (float*)d_out;

  const int B = in_sizes[1];                 // 4
  const int C = in_sizes[5];                 // 1024
  const int N = in_sizes[0] / (B * C);       // 2048
  const int M = B * N;                       // 8192

  // workspace carve-up (f16 buffers), 256B aligned
  char* ws = (char*)d_ws;
  auto alloc = [&](size_t bytes) {
    char* p = ws;
    ws += (bytes + 255) & ~(size_t)255;
    return p;
  };
  half_t* xh    = (half_t*)alloc((size_t)M * C * 2);
  half_t* Wqkvh = (half_t*)alloc((size_t)C * 3 * C * 2);
  half_t* Wprjh = (half_t*)alloc((size_t)C * C * 2);
  half_t* qkvh  = (half_t*)alloc((size_t)3 * M * C * 2);
  half_t* oh    = (half_t*)alloc((size_t)M * C * 2);
  half_t* uh    = (half_t*)alloc((size_t)M * C * 2);
  (void)ws_size;

  // K1: conversions
  {
    int n4;
    n4 = (M * C) / 4;
    cvt_f32_f16<<<(n4 + 255) / 256, 256, 0, stream>>>(x, xh, n4);
    n4 = (C * 3 * C) / 4;
    cvt_f32_f16<<<(n4 + 255) / 256, 256, 0, stream>>>(Wqkv, Wqkvh, n4);
    n4 = (C * C) / 4;
    cvt_f32_f16<<<(n4 + 255) / 256, 256, 0, stream>>>(Wproj, Wprjh, n4);
  }

  // K2: qkv GEMM  [M x C] @ [C x 3C] -> q/k/v f16
  {
    dim3 grid(3 * C / 64, M / 64);
    gemm_wmma<<<grid, 128, 0, stream>>>(xh, Wqkvh, bqkv, nullptr, qkvh,
                                        M, C, 3 * C, /*mode=*/1, B, N);
  }

  // K3: flash attention
  {
    dim3 grid(N / 64, 16, B);
    flash_attn<<<grid, 128, 0, stream>>>(qkvh, oh, B, N);
  }

  // K4: epilogue u = c_skip*x + c_out*o
  {
    int n = M * C;
    epilogue_u<<<(n + 255) / 256, 256, 0, stream>>>(x, oh, t, uh, B, n);
  }

  // K5: projection GEMM  [M x C] @ [C x C] + bproj -> fp32 out
  {
    dim3 grid(C / 64, M / 64);
    gemm_wmma<<<grid, 128, 0, stream>>>(uh, Wprjh, bproj, out, nullptr,
                                        M, C, C, /*mode=*/0, B, N);
  }
  (void)n_in; (void)out_size;
}
